// BoringAttention_22290880266294
// MI455X (gfx1250) — matrix operand order repaired and checked by reading, not compiled, not run on hardware
//
#include <hip/hip_runtime.h>
#include <hip/hip_bf16.h>
#include <math.h>
#include <stdint.h>

typedef __attribute__((ext_vector_type(16))) __bf16 v16bf;
typedef __attribute__((ext_vector_type(8)))  float  v8f;

constexpr int Bc = 4, Nc = 1024, Dc = 1024, Hc = 16, DHc = 64, MEMc = 16;
constexpr int JKc = Nc + MEMc;       // 1040
constexpr int INNERc = Hc * DHc;     // 1024
constexpr float SCALEc = 0.125f;     // 64^-0.5
constexpr float EPSc = 1e-6f;
constexpr float NEGc = -3.402823466e38f;

// ---------------------------------------------------------------------------
// Generic batched GEMM: C[M,N] = A[M,K] * Bt[N,K]^T   (bf16 inputs, f32 accum)
// One wave per 16x16 C tile; K stepped by 32 (v_wmma_f32_16x16x32_bf16),
// optional K%32==16 tail with zero-filled upper fragments.
// causalSkip: drop tiles that are entirely above the causal diagonal
// (n0 > m0+15+MEM) -- used for the Q*K^T logits GEMM.
// ---------------------------------------------------------------------------
__global__ void __launch_bounds__(256)
gemm_bf16_k(const __bf16* __restrict__ A, const __bf16* __restrict__ Bt,
            float* __restrict__ C,
            int M, int Nn, int K, int lda, int ldb, int ldc,
            int batches, int batchH,
            long sAb, long sAh, long sBb, long sBh, long sCb, long sCh,
            int causalSkip)
{
    const int mt = M >> 4, nt = Nn >> 4;
    const long tiles = (long)batches * mt * nt;
    const long tid = (long)blockIdx.x * blockDim.y + threadIdx.y;
    if (tid >= tiles) return;
    const int batch = (int)(tid / ((long)mt * nt));
    const int rem   = (int)(tid % ((long)mt * nt));
    const int m0 = (rem / nt) << 4;
    const int n0 = (rem % nt) << 4;
    if (causalSkip && n0 > m0 + 15 + MEMc) return;   // fully-masked logits tile

    const __bf16* Ab = A  + (batch / batchH) * sAb + (batch % batchH) * sAh;
    const __bf16* Bb = Bt + (batch / batchH) * sBb + (batch % batchH) * sBh;
    float*        Cb = C  + (batch / batchH) * sCb + (batch % batchH) * sCh;

    const int lane = threadIdx.x;
    const int r  = lane & 15;
    const int hi = lane >> 4;
    const __bf16* aRow = Ab + (long)(m0 + r) * lda;
    const __bf16* bRow = Bb + (long)(n0 + r) * ldb;

    v8f c;
#pragma unroll
    for (int i = 0; i < 8; ++i) c[i] = 0.f;

    const int kfull = K & ~31;
    for (int k = 0; k < kfull; k += 32) {
        v16bf a, b;
        uint4* ap = reinterpret_cast<uint4*>(&a);
        uint4* bp = reinterpret_cast<uint4*>(&b);
        // A 16x32 bf16 layout: lane holds K=[hi*8..hi*8+7] and [16+hi*8..23+hi*8]
        ap[0] = *reinterpret_cast<const uint4*>(aRow + k + hi * 8);
        ap[1] = *reinterpret_cast<const uint4*>(aRow + k + hi * 8 + 16);
        // B 32x16 bf16 layout: lane holds K=[hi*8..hi*8+7] and [16+hi*8..23+hi*8] (col = lane&15)
        bp[0] = *reinterpret_cast<const uint4*>(bRow + k + hi * 8);
        bp[1] = *reinterpret_cast<const uint4*>(bRow + k + hi * 8 + 16);
        c = __builtin_amdgcn_wmma_f32_16x16x32_bf16(false, a, false, b, (short)0, c, false, false);
    }
    if (K & 16) {  // tail: only K=k..k+15 valid
        const int k = kfull;
        uint4 zz; zz.x = zz.y = zz.z = zz.w = 0u;
        v16bf a, b;
        uint4* ap = reinterpret_cast<uint4*>(&a);
        uint4* bp = reinterpret_cast<uint4*>(&b);
        ap[0] = *reinterpret_cast<const uint4*>(aRow + k + hi * 8);
        ap[1] = zz;
        bp[0] = *reinterpret_cast<const uint4*>(bRow + k + hi * 8);
        bp[1] = zz;
        c = __builtin_amdgcn_wmma_f32_16x16x32_bf16(false, a, false, b, (short)0, c, false, false);
    }
    // C/D layout: VGPR v -> row m0+v+hi*8, col n0+(lane&15)
#pragma unroll
    for (int v = 0; v < 8; ++v)
        Cb[(long)(m0 + v + hi * 8) * ldc + n0 + r] = c[v];
}

// Same GEMM but A is fp32 in memory (converted to bf16 fragments in registers).
// causalK: cap K at round_up32(m0+16+MEM) -- attn is exactly zero beyond it.
__global__ void __launch_bounds__(256)
gemm_a32_k(const float* __restrict__ A, const __bf16* __restrict__ Bt,
           float* __restrict__ C,
           int M, int Nn, int K, int lda, int ldb, int ldc,
           int batches, int batchH,
           long sAb, long sAh, long sBb, long sBh, long sCb, long sCh,
           int causalK)
{
    const int mt = M >> 4, nt = Nn >> 4;
    const long tiles = (long)batches * mt * nt;
    const long tid = (long)blockIdx.x * blockDim.y + threadIdx.y;
    if (tid >= tiles) return;
    const int batch = (int)(tid / ((long)mt * nt));
    const int rem   = (int)(tid % ((long)mt * nt));
    const int m0 = (rem / nt) << 4;
    const int n0 = (rem % nt) << 4;
    if (causalK) {
        int kl = (m0 + 16 + MEMc + 31) & ~31;
        if (kl < K) K = kl;
    }
    const float*  Ab = A  + (batch / batchH) * sAb + (batch % batchH) * sAh;
    const __bf16* Bb = Bt + (batch / batchH) * sBb + (batch % batchH) * sBh;
    float*        Cb = C  + (batch / batchH) * sCb + (batch % batchH) * sCh;

    const int lane = threadIdx.x;
    const int r  = lane & 15;
    const int hi = lane >> 4;
    const float*  aRow = Ab + (long)(m0 + r) * lda;
    const __bf16* bRow = Bb + (long)(n0 + r) * ldb;

    v8f c;
#pragma unroll
    for (int i = 0; i < 8; ++i) c[i] = 0.f;

    const int kfull = K & ~31;
    for (int k = 0; k < kfull; k += 32) {
        float af[16];
        float4* fp = reinterpret_cast<float4*>(af);
        fp[0] = *reinterpret_cast<const float4*>(aRow + k + hi * 8);
        fp[1] = *reinterpret_cast<const float4*>(aRow + k + hi * 8 + 4);
        fp[2] = *reinterpret_cast<const float4*>(aRow + k + hi * 8 + 16);
        fp[3] = *reinterpret_cast<const float4*>(aRow + k + hi * 8 + 20);
        v16bf a, b;
#pragma unroll
        for (int i = 0; i < 16; ++i) a[i] = (__bf16)af[i];
        uint4* bp = reinterpret_cast<uint4*>(&b);
        bp[0] = *reinterpret_cast<const uint4*>(bRow + k + hi * 8);
        bp[1] = *reinterpret_cast<const uint4*>(bRow + k + hi * 8 + 16);
        c = __builtin_amdgcn_wmma_f32_16x16x32_bf16(false, a, false, b, (short)0, c, false, false);
    }
    if (K & 16) {
        const int k = kfull;
        float af[16];
        float4* fp = reinterpret_cast<float4*>(af);
        fp[0] = *reinterpret_cast<const float4*>(aRow + k + hi * 8);
        fp[1] = *reinterpret_cast<const float4*>(aRow + k + hi * 8 + 4);
#pragma unroll
        for (int i = 8; i < 16; ++i) af[i] = 0.f;
        v16bf a, b;
#pragma unroll
        for (int i = 0; i < 16; ++i) a[i] = (__bf16)af[i];
        uint4 zz; zz.x = zz.y = zz.z = zz.w = 0u;
        uint4* bp = reinterpret_cast<uint4*>(&b);
        bp[0] = *reinterpret_cast<const uint4*>(bRow + k + hi * 8);
        bp[1] = zz;
        c = __builtin_amdgcn_wmma_f32_16x16x32_bf16(false, a, false, b, (short)0, c, false, false);
    }
#pragma unroll
    for (int v = 0; v < 8; ++v)
        Cb[(long)(m0 + v + hi * 8) * ldc + n0 + r] = c[v];
}

// ---------------------------------------------------------------------------
// Elementwise / staging kernels
// ---------------------------------------------------------------------------
__global__ void cvt_f32_bf16_k(const float* __restrict__ in, __bf16* __restrict__ out, long n)
{
    long i = (long)blockIdx.x * blockDim.x + threadIdx.x;
    if (i < n) out[i] = (__bf16)in[i];
}

// Wt[n*K + k] = (bf16) W[k*Nn + n]
__global__ void transcvt_k(const float* __restrict__ W, __bf16* __restrict__ Wt, int K, int Nn)
{
    long i = (long)blockIdx.x * blockDim.x + threadIdx.x;
    if (i >= (long)K * Nn) return;
    int n = (int)(i / K);
    int k = (int)(i % K);
    Wt[i] = (__bf16)W[(long)k * Nn + n];
}

// l2-normalize 64-wide head rows of Q; qf[b,n,h*64+d] -> qb[b,h,n,d] bf16
__global__ void __launch_bounds__(256) post_q_k(const float* __restrict__ qf, __bf16* __restrict__ qb)
{
    int rowid = blockIdx.x * blockDim.y + threadIdx.y;
    if (rowid >= Bc * Nc * Hc) return;
    int h = rowid % Hc;
    int bn = rowid / Hc;
    int n = bn % Nc, b = bn / Nc;
    int lane = threadIdx.x, d0 = lane * 2;
    const float* src = qf + (long)bn * INNERc + h * DHc + d0;
    float x0 = src[0], x1 = src[1];
    float s = x0 * x0 + x1 * x1;
    for (int off = 16; off; off >>= 1) s += __shfl_xor(s, off, 32);
    float rn = rsqrtf(s + EPSc);
    __bf16* dst = qb + ((long)(b * Hc + h) * Nc + n) * DHc + d0;
    dst[0] = (__bf16)(x0 * rn);
    dst[1] = (__bf16)(x1 * rn);
}

// K: l2norm -> kb[b,h,MEM+n,d]; V: raw -> vT[b,h,d,MEM+n] (transposed for A*Bt^T)
__global__ void __launch_bounds__(256) post_kv_k(const float* __restrict__ kvf,
                                                 __bf16* __restrict__ kb, __bf16* __restrict__ vT)
{
    int rowid = blockIdx.x * blockDim.y + threadIdx.y;
    if (rowid >= Bc * Nc * Hc) return;
    int h = rowid % Hc;
    int bn = rowid / Hc;
    int n = bn % Nc, b = bn / Nc;
    int lane = threadIdx.x, d0 = lane * 2;
    const float* kf = kvf + (long)bn * (2 * INNERc) + h * DHc + d0;
    const float* vf = kf + INNERc;
    float k0 = kf[0], k1 = kf[1];
    float s = k0 * k0 + k1 * k1;
    for (int off = 16; off; off >>= 1) s += __shfl_xor(s, off, 32);
    float rn = rsqrtf(s + EPSc);
    __bf16* kdst = kb + ((long)(b * Hc + h) * JKc + MEMc + n) * DHc + d0;
    kdst[0] = (__bf16)(k0 * rn);
    kdst[1] = (__bf16)(k1 * rn);
    long vb = (long)(b * Hc + h) * DHc;
    vT[(vb + d0) * JKc + MEMc + n]     = (__bf16)vf[0];
    vT[(vb + d0 + 1) * JKc + MEMc + n] = (__bf16)vf[1];
}

// prepend learned memory slots (NOT normalized, per reference)
__global__ void memfill_k(const float* __restrict__ mem_k, const float* __restrict__ mem_v,
                          __bf16* __restrict__ kb, __bf16* __restrict__ vT)
{
    int i = blockIdx.x * blockDim.x + threadIdx.x;
    if (i >= Bc * Hc * MEMc * DHc) return;
    int d = i % DHc;
    int t = i / DHc;
    int m = t % MEMc; t /= MEMc;
    int h = t % Hc;
    int b = t / Hc;
    float kv = mem_k[(h * MEMc + m) * DHc + d];
    float vv = mem_v[(h * MEMc + m) * DHc + d];
    kb[((long)(b * Hc + h) * JKc + m) * DHc + d] = (__bf16)kv;
    vT[((long)(b * Hc + h) * DHc + d) * JKc + m] = (__bf16)vv;
}

// ---------------------------------------------------------------------------
// Fused talking-heads(pre) + scale + mask + softmax + talking-heads(post).
// One block per (b,i); the whole 16-head x 1040-key slice lives in LDS.
// The logits slice is pulled into LDS with GLOBAL_LOAD_ASYNC_TO_LDS_B128
// (ASYNCcnt-tracked, no VGPR round trip), one 16B granule per lane, skipping
// granules that are entirely causally masked (their LDS garbage is
// select-killed by the mask before softmax). Each wave issues its own
// s_wait_asynccnt 0 before the block barrier.
// ---------------------------------------------------------------------------
__global__ void __launch_bounds__(256)
mixsoft_k(const float* __restrict__ logits,
          const float* __restrict__ th_pre,
          const float* __restrict__ th_post,
          const unsigned char* __restrict__ mask,
          float* __restrict__ attn_out)
{
    extern __shared__ float smem[];
    float* s     = smem;               // [Hc][JKc]
    float* sPre  = smem + Hc * JKc;    // [256]
    float* sPost = sPre + Hc * Hc;     // [256]

    const int tid = threadIdx.x;
    const int bi = blockIdx.x;         // 0..B*N-1
    const int i = bi % Nc, b = bi / Nc;
    sPre[tid]  = th_pre[tid];          // Hc*Hc == 256 == blockDim
    sPost[tid] = th_post[tid];

    const long base = (long)b * Hc * Nc * JKc + (long)i * JKc;
    const long hs = (long)Nc * JKc;
    const int jmax = i + MEMc;         // causal-valid: j <= jmax

    // async global->LDS fill of the causal-valid region, 16B granules
    constexpr int GR = JKc / 4;        // 260 b128 granules per head row
    for (int gidx = tid; gidx < Hc * GR; gidx += 256) {
        int h = gidx / GR, jg = gidx - h * GR;
        int j0 = jg * 4;
        if (j0 <= jmax) {
            unsigned ldsAddr = (unsigned)(uintptr_t)(s + h * JKc + j0);
            const float* gp = logits + base + (long)h * hs + j0;
            asm volatile("global_load_async_to_lds_b128 %0, %1, off"
                         :: "v"(ldsAddr), "v"(gp) : "memory");
        }
    }
    asm volatile("s_wait_asynccnt 0x0" ::: "memory");
    __syncthreads();

    // column pass: pre-mix across heads, scale, mask (in place)
    for (int j = tid; j < JKc; j += 256) {
        bool ok = (j <= jmax) && (j < MEMc || mask[b * Nc + (j - MEMc)] != 0);
        float raw[Hc];
#pragma unroll
        for (int h = 0; h < Hc; ++h) raw[h] = s[h * JKc + j];
#pragma unroll
        for (int g = 0; g < Hc; ++g) {
            float acc = 0.f;
#pragma unroll
            for (int h = 0; h < Hc; ++h) acc += raw[h] * sPre[h * Hc + g];
            s[g * JKc + j] = ok ? acc * SCALEc : NEGc;
        }
    }
    __syncthreads();

    // row softmax per head (8 waves, 2 heads each)
    const int lane = tid & 31;
    const int wave = tid >> 5;
    for (int g = wave; g < Hc; g += 8) {
        float* row = s + g * JKc;
        float m = NEGc;
        for (int j = lane; j < JKc; j += 32) m = fmaxf(m, row[j]);
        for (int off = 16; off; off >>= 1) m = fmaxf(m, __shfl_xor(m, off, 32));
        float sum = 0.f;
        for (int j = lane; j < JKc; j += 32) { float e = __expf(row[j] - m); row[j] = e; sum += e; }
        for (int off = 16; off; off >>= 1) sum += __shfl_xor(sum, off, 32);
        float inv = 1.f / sum;
        for (int j = lane; j < JKc; j += 32) row[j] *= inv;
    }
    __syncthreads();

    // column pass: post-mix across heads, stream to global attn output
    for (int j = tid; j < JKc; j += 256) {
        float sv[Hc];
#pragma unroll
        for (int h = 0; h < Hc; ++h) sv[h] = s[h * JKc + j];
#pragma unroll
        for (int g = 0; g < Hc; ++g) {
            float acc = 0.f;
#pragma unroll
            for (int h = 0; h < Hc; ++h) acc += sv[h] * sPost[h * Hc + g];
            attn_out[base + g * hs + j] = acc;
        }
    }
}

// ---------------------------------------------------------------------------
extern "C" void kernel_launch(void* const* d_in, const int* in_sizes, int n_in,
                              void* d_out, int out_size, void* d_ws, size_t ws_size,
                              hipStream_t stream)
{
    (void)in_sizes; (void)n_in; (void)out_size; (void)ws_size;

    const float* x       = (const float*)d_in[0];
    const unsigned char* mask = (const unsigned char*)d_in[1];
    const float* Wq      = (const float*)d_in[2];
    const float* Wkv     = (const float*)d_in[3];
    const float* Wo      = (const float*)d_in[4];
    const float* mem_k   = (const float*)d_in[5];
    const float* mem_v   = (const float*)d_in[6];
    const float* th_pre  = (const float*)d_in[7];
    const float* th_post = (const float*)d_in[8];

    float* outFinal = (float*)d_out;                       // (B,N,D) fp32
    float* attnOut  = outFinal + (size_t)Bc * Nc * Dc;     // (B,H,N,JK) fp32

    char* ws = (char*)d_ws;
    size_t off = 0;
    auto carve = [&](size_t bytes) -> char* {
        char* p = ws + off;
        off += (bytes + 255) & ~(size_t)255;
        return p;
    };
    __bf16* wqT   = (__bf16*)carve((size_t)INNERc * Dc * 2);
    __bf16* wkvT  = (__bf16*)carve((size_t)2 * INNERc * Dc * 2);
    __bf16* woT   = (__bf16*)carve((size_t)Dc * INNERc * 2);
    __bf16* xb    = (__bf16*)carve((size_t)Bc * Nc * Dc * 2);
    float*  qf    = (float*) carve((size_t)Bc * Nc * INNERc * 4);
    float*  kvf   = (float*) carve((size_t)Bc * Nc * 2 * INNERc * 4);
    __bf16* qb    = (__bf16*)carve((size_t)Bc * Hc * Nc * DHc * 2);
    __bf16* kb    = (__bf16*)carve((size_t)Bc * Hc * JKc * DHc * 2);
    __bf16* vT    = (__bf16*)carve((size_t)Bc * Hc * DHc * JKc * 2);
    float*  logits= (float*) carve((size_t)Bc * Hc * Nc * JKc * 4);
    float*  outhF = (float*) carve((size_t)Bc * Nc * INNERc * 4);
    __bf16* outhB = (__bf16*)carve((size_t)Bc * Nc * INNERc * 2);

    const dim3 gw(32, 8);  // 8 waves per block, one 16x16 tile per wave

    // 1) stage inputs in bf16
    {
        long n = (long)Bc * Nc * Dc;
        cvt_f32_bf16_k<<<(int)((n + 255) / 256), 256, 0, stream>>>(x, xb, n);
    }
    transcvt_k<<<(Dc * INNERc + 255) / 256, 256, 0, stream>>>(Wq, wqT, Dc, INNERc);
    transcvt_k<<<(Dc * 2 * INNERc + 255) / 256, 256, 0, stream>>>(Wkv, wkvT, Dc, 2 * INNERc);
    transcvt_k<<<(INNERc * Dc + 255) / 256, 256, 0, stream>>>(Wo, woT, INNERc, Dc);

    // 2) projections: Q = x*Wq, KV = x*Wkv
    {
        long tiles = (long)(Bc * Nc / 16) * (INNERc / 16);
        gemm_bf16_k<<<(int)((tiles + 7) / 8), gw, 0, stream>>>(
            xb, wqT, qf, Bc * Nc, INNERc, Dc, Dc, Dc, INNERc,
            1, 1, 0, 0, 0, 0, 0, 0, 0);
    }
    {
        long tiles = (long)(Bc * Nc / 16) * (2 * INNERc / 16);
        gemm_bf16_k<<<(int)((tiles + 7) / 8), gw, 0, stream>>>(
            xb, wkvT, kvf, Bc * Nc, 2 * INNERc, Dc, Dc, Dc, 2 * INNERc,
            1, 1, 0, 0, 0, 0, 0, 0, 0);
    }

    // 3) qk-norm + head-major repack + memory slots
    {
        int rows = Bc * Nc * Hc;
        post_q_k<<<(rows + 7) / 8, gw, 0, stream>>>(qf, qb);
        post_kv_k<<<(rows + 7) / 8, gw, 0, stream>>>(kvf, kb, vT);
        memfill_k<<<(Bc * Hc * MEMc * DHc + 255) / 256, 256, 0, stream>>>(mem_k, mem_v, kb, vT);
    }

    // 4) logits = Q * K^T  (batched over b*h, K=64, causal tile skipping)
    {
        int batches = Bc * Hc;
        long tiles = (long)batches * (Nc / 16) * (JKc / 16);
        gemm_bf16_k<<<(int)((tiles + 7) / 8), gw, 0, stream>>>(
            qb, kb, logits, Nc, JKc, DHc, DHc, DHc, JKc,
            batches, Hc,
            (long)Hc * Nc * DHc,  (long)Nc * DHc,
            (long)Hc * JKc * DHc, (long)JKc * DHc,
            (long)Hc * Nc * JKc,  (long)Nc * JKc,
            1 /*causalSkip*/);
    }

    // 5) fused talking-heads pre + mask + softmax + talking-heads post
    {
        size_t smemBytes = (size_t)(Hc * JKc + 2 * Hc * Hc) * sizeof(float); // ~68.6KB
        mixsoft_k<<<Bc * Nc, 256, smemBytes, stream>>>(logits, th_pre, th_post, mask, attnOut);
    }

    // 6) out = attn * V  (fp32 attn read from d_out, causal K cap, K=1040 tail)
    {
        int batches = Bc * Hc;
        long tiles = (long)batches * (Nc / 16) * (DHc / 16);
        gemm_a32_k<<<(int)((tiles + 7) / 8), gw, 0, stream>>>(
            attnOut, vT, outhF, Nc, DHc, JKc, JKc, JKc, INNERc,
            batches, Hc,
            (long)Hc * Nc * JKc,  (long)Nc * JKc,
            (long)Hc * DHc * JKc, (long)DHc * JKc,
            (long)Nc * INNERc,    (long)DHc,
            1 /*causalK*/);
    }

    // 7) final projection: out @ Wo
    {
        long n = (long)Bc * Nc * INNERc;
        cvt_f32_bf16_k<<<(int)((n + 255) / 256), 256, 0, stream>>>(outhF, outhB, n);
        long tiles = (long)(Bc * Nc / 16) * (Dc / 16);
        gemm_bf16_k<<<(int)((tiles + 7) / 8), gw, 0, stream>>>(
            outhB, woT, outFinal, Bc * Nc, Dc, INNERc, INNERc, INNERc, Dc,
            1, 1, 0, 0, 0, 0, 0, 0, 0);
    }
}
